// GPTQOFTLinear_15659450761335
// MI455X (gfx1250) — matrix operand order repaired and checked by reading, not compiled, not run on hardware
//
#include <hip/hip_runtime.h>

// ---------------------------------------------------------------------------
// GPTQ-OFT linear for MI455X (gfx1250, wave32, WMMA, async-to-LDS)
//
//   y = x @ blockdiag(Q) @ W^T + b  ==  y = x @ W'^T + b,
//   W'[o, r*64+c] = sum_k W[o, r*64+k] * QT_r[k][c],  QT = (I-S)^{-1}(I+S)
//
// Pipeline (all on `stream`, every call):
//   1) cayley_kernel : 64 Gauss-Jordan solves (LDS)            -> QT      (ws)
//   2) wprime_kernel : fold rotation into W, split bf16 hi/lo  -> Whi/Wlo (ws)
//   3) xsplit_kernel : one-pass fp32 -> bf16 hi/lo split of x  -> Xhi/Xlo (ws)
//   4) gemm_kernel   : 8192x4096x4096 split-bf16 WMMA GEMM, double-buffered
//                      LDS fed by global_load_async_to_lds_b128 -> d_out
// Workspace: 1 + 32 + 32 + 64 + 64 = 193 MiB.
// ---------------------------------------------------------------------------

typedef __attribute__((ext_vector_type(8)))  float   f32x8;
typedef __attribute__((ext_vector_type(4)))  __bf16  bf16x4;
typedef __attribute__((ext_vector_type(8)))  __bf16  bf16x8;
typedef __attribute__((ext_vector_type(16))) __bf16  bf16x16;

#define IN_F   4096
#define OUT_F  4096
#define MTOT   8192      // BATCH * SEQ
#define BLK    64
#define RANKN  64

#define CAT16(a,b) __builtin_shufflevector((a),(b),0,1,2,3,4,5,6,7,8,9,10,11,12,13,14,15)

__device__ __forceinline__ f32x8 wmma_bf16(bf16x16 a, bf16x16 b, f32x8 c) {
  return __builtin_amdgcn_wmma_f32_16x16x32_bf16(false, a, false, b, (short)0, c, false, false);
}

// Async global -> LDS copy, 16 B per lane. lds = LDS byte address (addr[31:0]
// of the generic pointer), voff = 32-bit byte offset, base = uniform 64-bit
// base (SGPR pair), IMM = literal instruction offset (24-bit signed).
#define ASYNC_B128(lds, voff, base, IMM)                                      \
  asm volatile("global_load_async_to_lds_b128 %0, %1, %2 offset:" #IMM        \
               :: "v"(lds), "v"(voff), "s"(base) : "memory")

#define WAIT_ASYNC0() asm volatile("s_wait_asynccnt 0x0" ::: "memory")

// ---------------- Kernel 1: Cayley.  QT_r = (I - S)^{-1} (I + S) -------------
__global__ __launch_bounds__(256) void cayley_kernel(const float* __restrict__ R,
                                                     float* __restrict__ QT) {
  __shared__ float Ms[BLK * BLK];
  __shared__ float Xs[BLK * BLK];
  __shared__ float fac[BLK];
  __shared__ float pinv;
  const int r = blockIdx.x;
  const int tid = threadIdx.x;
  const float* Rb = R + (size_t)r * BLK * BLK;

  for (int idx = tid; idx < BLK * BLK; idx += 256) {
    int i = idx >> 6, j = idx & 63;
    float s  = 0.5f * (Rb[idx] - Rb[j * BLK + i]);   // skew
    float id = (i == j) ? 1.0f : 0.0f;
    Ms[idx] = id - s;                                // I - S
    Xs[idx] = id + s;                                // I + S  (becomes QT)
  }
  __syncthreads();

  for (int k = 0; k < BLK; ++k) {
    if (tid == 0) pinv = 1.0f / Ms[k * BLK + k];
    __syncthreads();
    float pv = pinv;
    if (tid < 64)       Ms[k * BLK + tid] *= pv;
    else if (tid < 128) Xs[k * BLK + (tid - 64)] *= pv;
    else if (tid < 192) fac[tid - 128] = Ms[(tid - 128) * BLK + k]; // fac[k] unused
    __syncthreads();
    for (int idx = tid; idx < BLK * 128; idx += 256) {
      int i = idx >> 7, c = idx & 127;
      if (i != k) {
        float f = fac[i];
        if (c < 64) Ms[i * BLK + c]        -= f * Ms[k * BLK + c];
        else        Xs[i * BLK + (c - 64)] -= f * Xs[k * BLK + (c - 64)];
      }
    }
    __syncthreads();
  }
  for (int idx = tid; idx < BLK * BLK; idx += 256)
    QT[(size_t)r * BLK * BLK + idx] = Xs[idx];
}

// ---------------- Kernel 2: W' = W * blockdiag(Q)^T, split into bf16 hi/lo ---
__global__ __launch_bounds__(256) void wprime_kernel(const float* __restrict__ W,
                                                     const float* __restrict__ QT,
                                                     __bf16* __restrict__ Whi,
                                                     __bf16* __restrict__ Wlo) {
  __shared__ float Qs[BLK * BLK];   // 16 KiB
  __shared__ float Ws[16 * BLK];    //  4 KiB
  const int r  = blockIdx.x;
  const int o0 = blockIdx.y * 16;
  const int tid = threadIdx.x;

  const float* Qr = QT + (size_t)r * BLK * BLK;
  for (int idx = tid; idx < BLK * BLK; idx += 256) Qs[idx] = Qr[idx];
  for (int idx = tid; idx < 16 * BLK; idx += 256) {
    int row = idx >> 6, col = idx & 63;
    Ws[idx] = W[(size_t)(o0 + row) * IN_F + r * BLK + col];
  }
  __syncthreads();

  const int c  = tid & 63;
  const int rg = tid >> 6;                 // 0..3
  for (int j = 0; j < 4; ++j) {
    int row = rg * 4 + j;
    float acc = 0.0f;
#pragma unroll 8
    for (int k = 0; k < BLK; ++k) acc += Ws[row * BLK + k] * Qs[k * BLK + c];
    __bf16 hi = (__bf16)acc;
    __bf16 lo = (__bf16)(acc - (float)hi);
    size_t off = (size_t)(o0 + row) * IN_F + r * BLK + c;
    Whi[off] = hi;
    Wlo[off] = lo;
  }
}

// ---------------- Kernel 3: one-pass split of x into bf16 hi/lo --------------
__global__ __launch_bounds__(256) void xsplit_kernel(const float* __restrict__ x,
                                                     __bf16* __restrict__ Xhi,
                                                     __bf16* __restrict__ Xlo) {
  // 33,554,432 floats = 8,388,608 float4; grid 4096 x 256 threads x 8 each.
  size_t t = (size_t)blockIdx.x * 256 + threadIdx.x;
#pragma unroll
  for (int i = 0; i < 8; ++i) {
    size_t f4 = t + (size_t)i * (4096u * 256u);
    const float4 v = ((const float4*)x)[f4];
    __bf16 h0 = (__bf16)v.x, h1 = (__bf16)v.y, h2 = (__bf16)v.z, h3 = (__bf16)v.w;
    bf16x4 hv = {h0, h1, h2, h3};
    bf16x4 lv = {(__bf16)(v.x - (float)h0), (__bf16)(v.y - (float)h1),
                 (__bf16)(v.z - (float)h2), (__bf16)(v.w - (float)h3)};
    ((bf16x4*)Xhi)[f4] = hv;
    ((bf16x4*)Xlo)[f4] = lv;
  }
}

// ---------------- Kernel 4: split-bf16 WMMA GEMM  y = x @ W'^T + b ----------
// 128x128 block tile, KT=32, 8 waves (2M x 4N), wave tile 64x32 (4x2 WMMAs).
// Double-buffered LDS filled by global_load_async_to_lds_b128 (ASYNCcnt).
#define BM     128
#define BN     128
#define KT     32
#define LSTR   40          // 32 + 8 pad: 80 B rows, 16B aligned, conflict-free
#define REG_AH 0
#define REG_AL (BM * LSTR * 2)           // 10240
#define REG_BH (2 * BM * LSTR * 2)       // 20480
#define REG_BL (3 * BM * LSTR * 2)       // 30720
#define BUFB   (4 * BM * LSTR * 2)       // 40960 bytes per buffer

__global__ __launch_bounds__(256) void gemm_kernel(const __bf16* __restrict__ Xhi,
                                                   const __bf16* __restrict__ Xlo,
                                                   const __bf16* __restrict__ Whi,
                                                   const __bf16* __restrict__ Wlo,
                                                   const float* __restrict__ bias,
                                                   float* __restrict__ out) {
  extern __shared__ __align__(16) char smem[];   // 2 * BUFB dynamic LDS

  const int tid   = threadIdx.x;
  const int n0    = blockIdx.x * BN;
  const int m0    = blockIdx.y * BM;
  const int wave  = tid >> 5;
  const int lane  = tid & 31;
  const int waveM = wave >> 2;             // 0..1
  const int waveN = wave & 3;              // 0..3
  const int lrow  = lane & 15;
  const int h     = lane >> 4;             // 0..1

  // Per-thread copy chunk: 512 b128 chunks per 128x32 region; thread t owns
  // chunk t (rows 0..63) and chunk t+256 (rows 64..127, +524288 B in global,
  // +5120 B in LDS via the literal instruction offset / LDS address).
  const int crow = tid >> 2;               // 0..63
  const int ccol = (tid & 3) * 8;          // 0,8,16,24 (bf16 elems)
  const unsigned gA = (unsigned)(((m0 + crow) * IN_F + ccol) * 2);  // bytes
  const unsigned gB = (unsigned)(((n0 + crow) * IN_F + ccol) * 2);
  const unsigned lchunk = (unsigned)((crow * LSTR + ccol) * 2);
  const unsigned lds0   = (unsigned)(uintptr_t)(void*)smem;         // LDS byte addr

  f32x8 acc[4][2] = {};

  auto stage = [&](int b, int kt) {
    const unsigned lb = lds0 + (unsigned)(b * BUFB) + lchunk;
    const unsigned ka = gA + (unsigned)(kt * 2);
    const unsigned kb = gB + (unsigned)(kt * 2);
    ASYNC_B128(lb + REG_AH,        ka, Xhi, 0);
    ASYNC_B128(lb + REG_AH + 5120, ka, Xhi, 524288);   // +64 rows
    ASYNC_B128(lb + REG_AL,        ka, Xlo, 0);
    ASYNC_B128(lb + REG_AL + 5120, ka, Xlo, 524288);
    ASYNC_B128(lb + REG_BH,        kb, Whi, 0);
    ASYNC_B128(lb + REG_BH + 5120, kb, Whi, 524288);
    ASYNC_B128(lb + REG_BL,        kb, Wlo, 0);
    ASYNC_B128(lb + REG_BL + 5120, kb, Wlo, 524288);
  };

  // Prologue: fill buffer 0 with tile kt=0.
  stage(0, 0);
  WAIT_ASYNC0();
  __syncthreads();

  for (int kt = 0; kt < IN_F; kt += KT) {
    const int par = (kt >> 5) & 1;
    const bool more = (kt + KT) < IN_F;
    if (more) stage(par ^ 1, kt + KT);     // async copy next tile (overlaps WMMA)

    const __bf16* AsH = (const __bf16*)(smem + par * BUFB + REG_AH);
    const __bf16* AsL = (const __bf16*)(smem + par * BUFB + REG_AL);
    const __bf16* BsH = (const __bf16*)(smem + par * BUFB + REG_BH);
    const __bf16* BsL = (const __bf16*)(smem + par * BUFB + REG_BL);

    // Fragments per ISA layouts (A and B: K = {8h..8h+7, 16+8h..})
    bf16x16 aH[4], aL[4], bH[2], bL[2];
#pragma unroll
    for (int m = 0; m < 4; ++m) {
      int row = waveM * 64 + m * 16 + lrow;
      const __bf16* p = &AsH[row * LSTR + 8 * h];
      aH[m] = CAT16(*(const bf16x8*)p, *(const bf16x8*)(p + 16));
      const __bf16* q = &AsL[row * LSTR + 8 * h];
      aL[m] = CAT16(*(const bf16x8*)q, *(const bf16x8*)(q + 16));
    }
#pragma unroll
    for (int n = 0; n < 2; ++n) {
      int col = waveN * 32 + n * 16 + lrow;
      const __bf16* p = &BsH[col * LSTR + 8 * h];
      bH[n] = CAT16(*(const bf16x8*)p, *(const bf16x8*)(p + 16));
      const __bf16* q = &BsL[col * LSTR + 8 * h];
      bL[n] = CAT16(*(const bf16x8*)q, *(const bf16x8*)(q + 16));
    }

    // Split-bf16 MAC: acc += hi*hi + hi*lo + lo*hi (fp32 accumulate)
#pragma unroll
    for (int m = 0; m < 4; ++m)
#pragma unroll
      for (int n = 0; n < 2; ++n) {
        acc[m][n] = wmma_bf16(aH[m], bH[n], acc[m][n]);
        acc[m][n] = wmma_bf16(aH[m], bL[n], acc[m][n]);
        acc[m][n] = wmma_bf16(aL[m], bH[n], acc[m][n]);
      }

    if (more) WAIT_ASYNC0();               // next tile landed in LDS
    __syncthreads();                       // single barrier per iteration
  }

  // Epilogue: C layout is reg i -> M = i + 8h, lane -> N = lrow.
#pragma unroll
  for (int m = 0; m < 4; ++m)
#pragma unroll
    for (int n = 0; n < 2; ++n) {
      int gn = n0 + waveN * 32 + n * 16 + lrow;
      float bv = bias[gn];
#pragma unroll
      for (int i = 0; i < 8; ++i) {
        int gm = m0 + waveM * 64 + m * 16 + 8 * h + i;
        out[(size_t)gm * OUT_F + gn] = acc[m][n][i] + bv;
      }
    }
}

// ---------------------------------------------------------------------------
extern "C" void kernel_launch(void* const* d_in, const int* in_sizes, int n_in,
                              void* d_out, int out_size, void* d_ws, size_t ws_size,
                              hipStream_t stream) {
  (void)in_sizes; (void)n_in; (void)out_size; (void)ws_size;
  const float* x     = (const float*)d_in[0];   // [2,4096,4096]
  const float* oft_r = (const float*)d_in[1];   // [64,64,64]
  const float* W     = (const float*)d_in[2];   // [4096,4096]
  const float* b     = (const float*)d_in[3];   // [4096]
  float* out = (float*)d_out;

  // Workspace layout (193 MiB total)
  float*  QT  = (float*)d_ws;                                    //  1 MiB
  __bf16* Whi = (__bf16*)((char*)d_ws + (size_t)(1u   << 20));   // 32 MiB
  __bf16* Wlo = (__bf16*)((char*)d_ws + (size_t)(33u  << 20));   // 32 MiB
  __bf16* Xhi = (__bf16*)((char*)d_ws + (size_t)(65u  << 20));   // 64 MiB
  __bf16* Xlo = (__bf16*)((char*)d_ws + (size_t)(129u << 20));   // 64 MiB

  cayley_kernel<<<RANKN, 256, 0, stream>>>(oft_r, QT);
  wprime_kernel<<<dim3(RANKN, OUT_F / 16), 256, 0, stream>>>(W, QT, Whi, Wlo);
  xsplit_kernel<<<4096, 256, 0, stream>>>(x, Xhi, Xlo);
  gemm_kernel<<<dim3(OUT_F / BN, MTOT / BM), 256, 2 * BUFB, stream>>>(Xhi, Xlo, Whi, Wlo, b, out);
}
